// LearnedDRoPEEnergy_7292854468685
// MI455X (gfx1250) — matrix operand order repaired and checked by reading, not compiled, not run on hardware
//
#include <hip/hip_runtime.h>
#include <math.h>

typedef __attribute__((ext_vector_type(16))) _Float16 v16h;
typedef __attribute__((ext_vector_type(8)))  _Float16 v8h;
typedef __attribute__((ext_vector_type(8)))  float    v8f;

#define B_ 16
#define K_ 256
#define H_ 64
#define W_ 64
#define KC 64          // K-chunk staged in LDS
#define KSTR 72        // padded per-pixel k-stride (halfs): bank stride 36 dwords -> conflict-free b128

// ---------------- pass 0: w = softplus(w_logit), fp32 + fp16 ----------------
__global__ void prep_kernel(const float* __restrict__ wl,
                            float* __restrict__ w32, _Float16* __restrict__ w16) {
    int k = threadIdx.x;              // 256 threads
    float x = wl[k];
    float w = (x > 20.f) ? x : log1pf(expf(x));
    w32[k] = w;
    w16[k] = (_Float16)w;
}

// ---------------- pass 1: s(j) = sum_k w_k * z_k(j)  (single pass over z) ----------------
__global__ void s_kernel(const float* __restrict__ z, const float* __restrict__ w32,
                         float* __restrict__ s) {
    __shared__ float ws[K_];
    ws[threadIdx.x] = w32[threadIdx.x];
    __syncthreads();
    int j   = blockIdx.x * 256 + threadIdx.x;     // 65536 pixels total
    int b   = j >> 12;
    int rem = j & 4095;                           // y*64+x
    const float* zp = z + (size_t)b * K_ * 4096 + rem;
    float acc = 0.f;
#pragma unroll 4
    for (int k = 0; k < K_; ++k)
        acc = fmaf(zp[(size_t)k * 4096], ws[k], acc);
    s[j] = acc;
}

// ---------------- pass 2: cross terms via WMMA, gate, per-wave partial ----------------
__global__ __launch_bounds__(128) void energy_kernel(const float* __restrict__ z,
                                                     const _Float16* __restrict__ w16,
                                                     const float* __restrict__ s,
                                                     const float* __restrict__ tau_p,
                                                     float* __restrict__ partials) {
    // rows y..y+2, one 64-K chunk, fp16, [row][x][k] with padded k-stride
    __shared__ _Float16 lds[3 * 64 * KSTR];       // 13824 halfs = 27 KB (3*64*72*2 = 27,648 B)

    const int tid  = threadIdx.x;
    const int lane = tid & 31;
    const int wave = tid >> 5;                    // 4 waves, 16 pixels each
    const int blk  = blockIdx.x;                  // b*64 + y
    const int b    = blk >> 6;
    const int y    = blk & 63;
    const int x0   = wave * 16;
    const float tau = tau_p[0];

    // canonical half-offsets (dy >= 0); result doubled by torus symmetry
    const int dy[8] = {1, 0, 1, 1, 2, 0, 2, 2};
    const int dx[8] = {0, 1, 1,-1, 0, 2, 2,-2};

    v8f cacc[8] = {};                             // per-offset 16x16 f32 accumulators

    const int n  = lane & 15;
    const int hi = lane >> 4;                     // 0 or 1

    for (int kc = 0; kc < 4; ++kc) {
        __syncthreads();
        // stage rows y..y+2, k in [kc*64, kc*64+64) -> fp16 LDS (coalesced global reads)
        for (int i = tid; i < 3 * 64 * KC; i += 128) {
            int x  = i & 63;
            int kk = (i >> 6) & 63;
            int r  = i >> 12;                     // 0..2
            int ys = (y + r) & 63;
            float v = z[(((size_t)b * K_ + (kc * KC + kk)) * H_ + ys) * W_ + x];
            lds[(r * 64 + x) * KSTR + kk] = (_Float16)v;
        }
        __syncthreads();

        // A fragments (w .* z at center row) for the 2 k-steps of this chunk
        v16h afrag[2];
#pragma unroll
        for (int ks = 0; ks < 2; ++ks) {
            int koffA = ks * 32 + hi * 8;         // ISA 16-bit A layout (16x32)
            const _Float16* pz = &lds[(0 * 64 + (x0 + n)) * KSTR + koffA];
            v8h zlo = *(const v8h*)(pz);
            v8h zhi = *(const v8h*)(pz + 16);
            const _Float16* pw = &w16[kc * KC + koffA];
            v8h wlo = *(const v8h*)(pw);
            v8h whi = *(const v8h*)(pw + 16);
            v16h zf = __builtin_shufflevector(zlo, zhi, 0,1,2,3,4,5,6,7,8,9,10,11,12,13,14,15);
            v16h wf = __builtin_shufflevector(wlo, whi, 0,1,2,3,4,5,6,7,8,9,10,11,12,13,14,15);
            afrag[ks] = zf * wf;                  // packed fp16 mul
        }

        // 8 offsets x 2 k-steps of v_wmma_f32_16x16x32_f16, chained C->D
#pragma unroll
        for (int o = 0; o < 8; ++o) {
            int r  = dy[o];
            int xn = (x0 + n + dx[o] + 64) & 63;
#pragma unroll
            for (int ks = 0; ks < 2; ++ks) {
                int koffB = ks * 32 + hi * 8;     // ISA 16-bit B layout (32x16)
                const _Float16* pb = &lds[(r * 64 + xn) * KSTR + koffB];
                v8h blo = *(const v8h*)(pb);
                v8h bhi = *(const v8h*)(pb + 16);
                v16h bf = __builtin_shufflevector(blo, bhi, 0,1,2,3,4,5,6,7,8,9,10,11,12,13,14,15);
                cacc[o] = __builtin_amdgcn_wmma_f32_16x16x32_f16(
                    false, afrag[ks], false, bf, (short)0, cacc[o], false, false);
            }
        }
    }

    // diagonal of each 16x16 product lives in lanes 0-7 (M=0..7) and 24-31 (M=8..15)
    float esum = 0.f;
    bool diag = (lane < 8) || (lane >= 24);
    if (diag) {
        int m  = (lane < 8) ? lane : (lane - 16);
        int v  = lane & 7;                        // accumulator element holding the diagonal
        int xm = x0 + m;
        float s_c = s[((size_t)b * 64 + y) * 64 + xm];
#pragma unroll
        for (int o = 0; o < 8; ++o) {
            float c = cacc[o][0];
#pragma unroll
            for (int i = 1; i < 8; ++i) if (v == i) c = cacc[o][i];
            int yn  = (y + dy[o]) & 63;
            int xn2 = (xm + dx[o] + 64) & 63;
            float s_n  = s[((size_t)b * 64 + yn) * 64 + xn2];
            float dist = s_c + s_n - 2.f * c;     // weighted Hamming distance
            float gate = 1.f / (1.f + expf(dist - tau));
            esum += gate * dist;
        }
    }
    esum *= 2.f;                                  // each +/- offset pair counted once

    // wave reduction (wave32), one partial per wave
    for (int off2 = 16; off2 >= 1; off2 >>= 1)
        esum += __shfl_xor(esum, off2, 32);
    if (lane == 0) partials[blk * 4 + wave] = esum;
}

// ---------------- pass 3: deterministic reduction of 256 partials per batch ----------------
__global__ void reduce_kernel(const float* __restrict__ partials, float* __restrict__ out) {
    __shared__ float buf[256];
    int b = blockIdx.x;
    buf[threadIdx.x] = partials[b * 256 + threadIdx.x];
    __syncthreads();
    for (int st = 128; st >= 1; st >>= 1) {
        if (threadIdx.x < st) buf[threadIdx.x] += buf[threadIdx.x + st];
        __syncthreads();
    }
    if (threadIdx.x == 0) out[b] = buf[0];
}

extern "C" void kernel_launch(void* const* d_in, const int* in_sizes, int n_in,
                              void* d_out, int out_size, void* d_ws, size_t ws_size,
                              hipStream_t stream) {
    const float* z   = (const float*)d_in[0];     // (16,256,64,64) fp32 binary
    const float* wl  = (const float*)d_in[1];     // (256,)
    const float* tau = (const float*)d_in[2];     // scalar
    float* out = (float*)d_out;                   // (16,)

    char* ws = (char*)d_ws;
    float*    w32      = (float*)(ws);                       // 1 KB
    _Float16* w16      = (_Float16*)(ws + 1024);             // 0.5 KB
    float*    sbuf     = (float*)(ws + 2048);                // 256 KB: s(b,y,x)
    float*    partials = (float*)(ws + 2048 + 65536 * 4);    // 16 KB: 1024 blk x 4 waves

    prep_kernel  <<<1,    256, 0, stream>>>(wl, w32, w16);
    s_kernel     <<<256,  256, 0, stream>>>(z, w32, sbuf);
    energy_kernel<<<1024, 128, 0, stream>>>(z, w16, sbuf, tau, partials);
    reduce_kernel<<<16,   256, 0, stream>>>(partials, out);
}
